// lstm_multi_variable_66443144069382
// MI455X (gfx1250) — matrix operand order repaired and checked by reading, not compiled, not run on hardware
//
#include <hip/hip_runtime.h>

#define HSZ  16
#define TLEN 512

typedef __attribute__((ext_vector_type(16))) _Float16 v16h;
typedef __attribute__((ext_vector_type(8)))  float    v8f;

__device__ __forceinline__ float rcp_(float x)   { return __builtin_amdgcn_rcpf(x); }
__device__ __forceinline__ float sigm_(float x)  { return rcp_(1.f + __expf(-x)); }
__device__ __forceinline__ float tanh_(float x)  { return 1.f - 2.f * rcp_(__expf(2.f * x) + 1.f); }
__device__ __forceinline__ float elu_(float x)   { return x > 0.f ? x : __expf(x) - 1.f; }

// ---------------------------------------------------------------------------
// Kernel 1: layer-0 bidirectional LSTM (input dim 1).
// grid = (B/16, 2 dirs), block = 1 wave.  Recurrence h@Whh.T via
// v_wmma_f32_16x16x32_f16 (M=16 batch, N=16 per gate chunk i/f/g/o,
// K=16 zero-padded to 32).  Double-buffered h in LDS, stride-stepped
// pointers for x reads and o0 stores.
// ---------------------------------------------------------------------------
__global__ __launch_bounds__(32)
void lstm_l0_kernel(const float* __restrict__ x,
                    const float* __restrict__ WihF, const float* __restrict__ WhhF,
                    const float* __restrict__ bihF, const float* __restrict__ bhhF,
                    const float* __restrict__ WihB, const float* __restrict__ WhhB,
                    const float* __restrict__ bihB, const float* __restrict__ bhhB,
                    _Float16* __restrict__ o0, int Btot)
{
    __shared__ float    xs[16 * TLEN];          // 32 KB staged x tile
    __shared__ _Float16 hbuf[2][16 * HSZ];      // ping-pong h state, [row][hidden]

    const int bt   = blockIdx.x;
    const int dir  = blockIdx.y;
    const int lane = threadIdx.x;
    const int nn   = lane & 15;
    const int hi   = (lane >> 4) & 1;
    const int rb   = hi * 8;                    // row base for this lane half (C/D layout)

    const float* Wih = dir ? WihB : WihF;
    const float* Whh = dir ? WhhB : WhhF;
    const float* bih = dir ? bihB : bihF;
    const float* bhh = dir ? bhhB : bhhF;

    float wih_s[4], bias_s[4];
    v16h  Bm[4];
    #pragma unroll
    for (int c = 0; c < 4; ++c) {
        const int g = c * 16 + nn;
        wih_s[c]  = Wih[g];                     // Wih0 is [64,1]
        bias_s[c] = bih[g] + bhh[g];
        // B layout (16-bit 32x16): e<8 -> K=e+8*hi, e>=8 -> K>=16 (zero pad), N=nn
        #pragma unroll
        for (int e = 0; e < 16; ++e)
            Bm[c][e] = (_Float16)(e < 8 ? Whh[g * HSZ + 8 * hi + e] : 0.f);
    }

    const float* xrow = x + (size_t)(bt * 16) * TLEN;
    for (int i = lane; i < 16 * TLEN; i += 32) xs[i] = xrow[i];
    for (int i = lane; i < 16 * HSZ;  i += 32) hbuf[0][i] = (_Float16)0.f;

    float cst[8];
    #pragma unroll
    for (int j = 0; j < 8; ++j) cst[j] = 0.f;

    const int       t0      = dir ? (TLEN - 1) : 0;
    const int       tstep   = dir ? -1 : 1;
    const long long ostride = (long long)tstep * Btot * 32;     // halfs per time step
    const float*    xp      = xs + rb * TLEN + t0;
    _Float16*       op      = o0 + ((size_t)t0 * Btot + bt * 16 + rb) * 32 + dir * 16 + nn;

    for (int s = 0; s < TLEN; ++s) {
        const int cur = s & 1, nxt = cur ^ 1;
        __syncthreads();
        // A layout (16-bit 16x32): M=nn, e<8 -> K=e+8*hi, e>=8 -> K>=16 (zero pad)
        const _Float16* hp = &hbuf[cur][nn * HSZ + 8 * hi];
        v16h A;
        #pragma unroll
        for (int e = 0; e < 8; ++e)  A[e] = hp[e];
        #pragma unroll
        for (int e = 8; e < 16; ++e) A[e] = (_Float16)0.f;

        float xv[8];
        #pragma unroll
        for (int j = 0; j < 8; ++j) xv[j] = xp[j * TLEN];

        v8f D[4];
        #pragma unroll
        for (int c = 0; c < 4; ++c) {
            v8f acc;                                        // C = x*Wih + bias
            #pragma unroll
            for (int j = 0; j < 8; ++j) acc[j] = xv[j] * wih_s[c] + bias_s[c];
            D[c] = __builtin_amdgcn_wmma_f32_16x16x32_f16(false, A, false, Bm[c],
                                                          (short)0, acc, false, false);
        }
        #pragma unroll
        for (int j = 0; j < 8; ++j) {
            const float iv = sigm_(D[0][j]);
            const float fv = sigm_(D[1][j]);
            const float gv = tanh_(D[2][j]);
            const float ov = sigm_(D[3][j]);
            cst[j] = fv * cst[j] + iv * gv;
            const _Float16 hh = (_Float16)(ov * tanh_(cst[j]));
            hbuf[nxt][(rb + j) * HSZ + nn] = hh;            // next step's A source
            op[j * 32] = hh;
        }
        xp += tstep;
        op += ostride;
    }
}

// ---------------------------------------------------------------------------
// Kernel 2: layer-1 LSTM, one direction per block (grid = (B/16, 2)).
// Per step: 4 WMMAs for x@Wih1.T (full K=32) chained through C into
// 4 WMMAs for h@Whh1.T.  Ain is software-pipelined (prefetched one step
// ahead so the o0 load is off the recurrence critical path).  Final h
// captured from registers into featbuf ( [b][ hT_b(16) | hT_f(16) ] f32 ).
// ---------------------------------------------------------------------------
__global__ __launch_bounds__(32)
void lstm_l1_kernel(const _Float16* __restrict__ o0,
                    const float* __restrict__ WihF, const float* __restrict__ WhhF,
                    const float* __restrict__ bihF, const float* __restrict__ bhhF,
                    const float* __restrict__ WihB, const float* __restrict__ WhhB,
                    const float* __restrict__ bihB, const float* __restrict__ bhhB,
                    float* __restrict__ featbuf, int Btot)
{
    __shared__ _Float16 hbuf[2][16 * HSZ];

    const int bt   = blockIdx.x;
    const int dir  = blockIdx.y;                // 0 = forward, 1 = backward
    const int lane = threadIdx.x;
    const int nn   = lane & 15;
    const int hi   = (lane >> 4) & 1;
    const int rb   = hi * 8;

    const float* Wih = dir ? WihB : WihF;
    const float* Whh = dir ? WhhB : WhhF;
    const float* bih = dir ? bihB : bihF;
    const float* bhh = dir ? bhhB : bhhF;

    float bias_s[4];
    v16h  Bin[4], Bhh[4];
    #pragma unroll
    for (int c = 0; c < 4; ++c) {
        const int g = c * 16 + nn;
        bias_s[c] = bih[g] + bhh[g];
        #pragma unroll
        for (int e = 0; e < 16; ++e) {
            Bin[c][e] = (_Float16)Wih[g * 32 + (e < 8 ? 8 * hi + e : 16 + 8 * hi + (e - 8))];        // e<8 -> K=e+8*hi, e>=8 -> 16+(e-8)+8*hi
            Bhh[c][e] = (_Float16)(e < 8 ? Whh[g * HSZ + 8 * hi + e] : 0.f);    // K=16 padded
        }
    }

    for (int i = lane; i < 16 * HSZ; i += 32) hbuf[0][i] = (_Float16)0.f;
    float cst[8], hout[8];
    #pragma unroll
    for (int j = 0; j < 8; ++j) { cst[j] = 0.f; hout[j] = 0.f; }

    const int       t0      = dir ? (TLEN - 1) : 0;
    const long long istride = (long long)(dir ? -1 : 1) * Btot * 32;    // halfs per step
    const _Float16* ip      = o0 + ((size_t)t0 * Btot + bt * 16 + nn) * 32;

    // Prologue: load Ain for step 0.
    // A from o0: row M=nn, features K=0..31 (e<8 -> K=e+8*hi, e>=8 -> 16+(e-8)+8*hi)
    v16h Ain;
    #pragma unroll
    for (int e = 0; e < 8; ++e)  Ain[e] = ip[8 * hi + e];
    #pragma unroll
    for (int e = 8; e < 16; ++e) Ain[e] = ip[16 + 8 * hi + (e - 8)];

    for (int s = 0; s < TLEN; ++s) {
        const int cur = s & 1, nxt = cur ^ 1;
        __syncthreads();
        const _Float16* hp = &hbuf[cur][nn * HSZ + 8 * hi];
        v16h Ah;
        #pragma unroll
        for (int e = 0; e < 8; ++e)  Ah[e] = hp[e];
        #pragma unroll
        for (int e = 8; e < 16; ++e) Ah[e] = (_Float16)0.f;

        // Prefetch next step's Ain (independent of h -> overlaps WMMA/gates/barrier).
        v16h AinN = Ain;
        if (s + 1 < TLEN) {                     // scalar-uniform branch, EXEC untouched
            ip += istride;
            #pragma unroll
            for (int e = 0; e < 8; ++e)  AinN[e] = ip[8 * hi + e];
            #pragma unroll
            for (int e = 8; e < 16; ++e) AinN[e] = ip[16 + 8 * hi + (e - 8)];
        }

        v8f D[4];
        #pragma unroll
        for (int c = 0; c < 4; ++c) {
            v8f acc;
            #pragma unroll
            for (int j = 0; j < 8; ++j) acc[j] = bias_s[c];
            acc  = __builtin_amdgcn_wmma_f32_16x16x32_f16(false, Ain, false, Bin[c],
                                                          (short)0, acc, false, false);
            D[c] = __builtin_amdgcn_wmma_f32_16x16x32_f16(false, Ah,  false, Bhh[c],
                                                          (short)0, acc, false, false);
        }
        #pragma unroll
        for (int j = 0; j < 8; ++j) {
            const float iv = sigm_(D[0][j]);
            const float fv = sigm_(D[1][j]);
            const float gv = tanh_(D[2][j]);
            const float ov = sigm_(D[3][j]);
            cst[j] = fv * cst[j] + iv * gv;
            const float hv = ov * tanh_(cst[j]);
            hout[j] = hv;
            hbuf[nxt][(rb + j) * HSZ + nn] = (_Float16)hv;
        }
        Ain = AinN;
    }

    // feat = concat(hT_b, hT_f): backward -> cols 0..15, forward -> cols 16..31
    const int off = dir ? 0 : 16;
    #pragma unroll
    for (int j = 0; j < 8; ++j)
        featbuf[(size_t)(bt * 16 + rb + j) * 32 + off + nn] = hout[j];
}

// ---------------------------------------------------------------------------
// Kernel 3: ELU/MLP head, one thread per batch row.
// ---------------------------------------------------------------------------
__global__ __launch_bounds__(128)
void head_kernel(const float* __restrict__ featbuf,
                 const float* __restrict__ W1, const float* __restrict__ b1,
                 const float* __restrict__ W2, const float* __restrict__ b2,
                 float* __restrict__ out, int Btot)
{
    const int b = blockIdx.x * blockDim.x + threadIdx.x;
    if (b >= Btot) return;
    float zin[32];
    #pragma unroll
    for (int k = 0; k < 32; ++k) zin[k] = elu_(featbuf[(size_t)b * 32 + k]);
    float z1v[25];
    for (int o = 0; o < 25; ++o) {
        float sacc = b1[o];
        #pragma unroll
        for (int k = 0; k < 32; ++k) sacc += zin[k] * W1[o * 32 + k];
        z1v[o] = elu_(sacc);
    }
    for (int o = 0; o < 20; ++o) {
        float sacc = b2[o];
        #pragma unroll
        for (int k = 0; k < 25; ++k) sacc += z1v[k] * W2[o * 25 + k];
        out[(size_t)b * 20 + o] = elu_(sacc);
    }
}

extern "C" void kernel_launch(void* const* d_in, const int* in_sizes, int n_in,
                              void* d_out, int out_size, void* d_ws, size_t ws_size,
                              hipStream_t stream)
{
    (void)n_in; (void)out_size; (void)ws_size;
    const float* x     = (const float*)d_in[0];
    const float* Wih0f = (const float*)d_in[1];
    const float* Whh0f = (const float*)d_in[2];
    const float* bih0f = (const float*)d_in[3];
    const float* bhh0f = (const float*)d_in[4];
    const float* Wih0b = (const float*)d_in[5];
    const float* Whh0b = (const float*)d_in[6];
    const float* bih0b = (const float*)d_in[7];
    const float* bhh0b = (const float*)d_in[8];
    const float* Wih1f = (const float*)d_in[9];
    const float* Whh1f = (const float*)d_in[10];
    const float* bih1f = (const float*)d_in[11];
    const float* bhh1f = (const float*)d_in[12];
    const float* Wih1b = (const float*)d_in[13];
    const float* Whh1b = (const float*)d_in[14];
    const float* bih1b = (const float*)d_in[15];
    const float* bhh1b = (const float*)d_in[16];
    const float* W1    = (const float*)d_in[17];
    const float* b1    = (const float*)d_in[18];
    const float* W2    = (const float*)d_in[19];
    const float* b2    = (const float*)d_in[20];
    float* out = (float*)d_out;

    const int Btot = in_sizes[0] / TLEN;            // x is [B, T] with T = 512

    // Workspace layout: o0 f16 [T][B][32], then featbuf f32 [B][32].
    _Float16* o0      = (_Float16*)d_ws;
    float*    featbuf = (float*)((char*)d_ws + (size_t)TLEN * Btot * 32 * sizeof(_Float16));

    dim3 g01(Btot / 16, 2);
    lstm_l0_kernel<<<g01, 32, 0, stream>>>(x, Wih0f, Whh0f, bih0f, bhh0f,
                                           Wih0b, Whh0b, bih0b, bhh0b, o0, Btot);

    lstm_l1_kernel<<<g01, 32, 0, stream>>>(o0,
        Wih1f, Whh1f, bih1f, bhh1f, Wih1b, Whh1b, bih1b, bhh1b, featbuf, Btot);

    head_kernel<<<(Btot + 127) / 128, 128, 0, stream>>>(featbuf, W1, b1, W2, b2, out, Btot);
}
